// PerceiverAttention_16320875724839
// MI455X (gfx1250) — compile-verified, matrix operand order checked by reading, not run on hardware
//
#include <hip/hip_runtime.h>
#include <hip/hip_bf16.h>

// ---------------------------------------------------------------------------
// Types for CDNA5 WMMA (gfx1250, wave32)
// ---------------------------------------------------------------------------
typedef __bf16 bf16_t;
typedef __attribute__((ext_vector_type(8)))  __bf16 v8bf;
typedef __attribute__((ext_vector_type(16))) __bf16 v16bf;
typedef __attribute__((ext_vector_type(8)))  float  v8f;
typedef __attribute__((ext_vector_type(4)))  float  v4f;

union Frag16 {
    v16bf v;
    v8bf  h[2];
};

static __device__ inline v8f wmma_bf16(v16bf a, v16bf b, v8f c) {
    // D = A(16x32 bf16) x B(32x16 bf16) + C(16x16 f32)
    return __builtin_amdgcn_wmma_f32_16x16x32_bf16(
        /*neg_a=*/false, a, /*neg_b=*/false, b,
        /*c_mod=*/(short)0, c, /*reuse_a=*/false, /*reuse_b=*/false);
}

// ---- CDNA5 async global->LDS copy (bypasses VGPRs, tracked by ASYNCcnt) ----
#ifndef __has_builtin
#define __has_builtin(x) 0
#endif

#if __has_builtin(__builtin_amdgcn_global_load_async_to_lds_b128)
#define HAVE_ASYNC_LDS 1
#else
#define HAVE_ASYNC_LDS 0
#endif

// Exact pointee type from the builtin's signature (clang diagnostic):
//   'int __attribute__((__vector_size__(4 * sizeof(int))))'
typedef int v4i_raw __attribute__((vector_size(4 * sizeof(int))));
typedef __attribute__((address_space(1))) v4i_raw g1_v4i;   // global
typedef __attribute__((address_space(3))) v4i_raw l3_v4i;   // LDS

static __device__ inline void pa_async_copy_b128(const void* g, void* l) {
#if HAVE_ASYNC_LDS
    __builtin_amdgcn_global_load_async_to_lds_b128(
        (g1_v4i*)g, (l3_v4i*)l, /*offset=*/0, /*cpol=*/0);
#else
    (void)g; (void)l;
#endif
}

static __device__ inline void pa_async_wait0() {
#if HAVE_ASYNC_LDS
#if __has_builtin(__builtin_amdgcn_s_wait_asynccnt)
    __builtin_amdgcn_s_wait_asynccnt(0);
#else
    asm volatile("s_wait_asynccnt 0x0" ::: "memory");
#endif
#endif
}

// Problem constants
#define B_SZ     8
#define N_MEDIA  4096
#define N_LAT    64
#define DIM_     1024
#define HEADS_   16
#define DHEAD_   64
#define HIDDEN_  1024
#define CTX_LEN  (N_MEDIA + N_LAT)   // 4160

// ---------------------------------------------------------------------------
// Kernel 1: fused LayerNorm of x and latents -> bf16 ctx[B,4160,D] (+ compact
// latent copy for the q projection). One 128-thread block per row.
// ---------------------------------------------------------------------------
__global__ __launch_bounds__(128)
void pa_layernorm_kernel(const float* __restrict__ x,
                         const float* __restrict__ latents,
                         const float* __restrict__ g_media,
                         const float* __restrict__ b_media,
                         const float* __restrict__ g_lat,
                         const float* __restrict__ b_lat,
                         bf16_t* __restrict__ ctx,
                         bf16_t* __restrict__ lnc)
{
    const int row = blockIdx.x;            // [0, B*4160)
    const int b   = row / CTX_LEN;
    const int s   = row - b * CTX_LEN;
    const bool is_lat = (s >= N_MEDIA);

    const float* src;
    const float *g, *bb;
    if (!is_lat) {
        src = x + ((size_t)b * N_MEDIA + s) * DIM_;
        g = g_media; bb = b_media;
    } else {
        src = latents + ((size_t)b * N_LAT + (s - N_MEDIA)) * DIM_;
        g = g_lat; bb = b_lat;
    }

    const int t = threadIdx.x;
    const v4f* s4 = (const v4f*)src;
    v4f a0 = s4[2 * t + 0];
    v4f a1 = s4[2 * t + 1];

    float s1 = 0.f, s2 = 0.f;
#pragma unroll
    for (int i = 0; i < 4; ++i) { s1 += a0[i]; s2 += a0[i] * a0[i]; }
#pragma unroll
    for (int i = 0; i < 4; ++i) { s1 += a1[i]; s2 += a1[i] * a1[i]; }

#pragma unroll
    for (int msk = 16; msk >= 1; msk >>= 1) {
        s1 += __shfl_xor(s1, msk, 32);
        s2 += __shfl_xor(s2, msk, 32);
    }
    __shared__ float redm[4], redv[4];
    if ((t & 31) == 0) { redm[t >> 5] = s1; redv[t >> 5] = s2; }
    __syncthreads();
    const float ts1 = redm[0] + redm[1] + redm[2] + redm[3];
    const float ts2 = redv[0] + redv[1] + redv[2] + redv[3];
    const float mean = ts1 * (1.0f / DIM_);
    const float var  = ts2 * (1.0f / DIM_) - mean * mean;   // ddof=0, as jnp.var
    const float rs   = rsqrtf(var + 1e-5f);

    v8bf y8;
#pragma unroll
    for (int i = 0; i < 8; ++i) {
        const float xv = (i < 4) ? a0[i] : a1[i - 4];
        const int col  = t * 8 + i;
        const float y  = (xv - mean) * rs * g[col] + bb[col];
        y8[i] = (bf16_t)y;
    }
    *(v8bf*)(ctx + (size_t)row * DIM_ + t * 8) = y8;
    if (is_lat)
        *(v8bf*)(lnc + ((size_t)b * N_LAT + (s - N_MEDIA)) * DIM_ + t * 8) = y8;
}

// ---------------------------------------------------------------------------
// Kernel 2: fp32 -> bf16 conversion (weights)
// ---------------------------------------------------------------------------
__global__ void pa_f32_to_bf16_kernel(const float* __restrict__ in,
                                      bf16_t* __restrict__ out, size_t n)
{
    size_t i = (size_t)blockIdx.x * blockDim.x + threadIdx.x;
    const size_t stride = (size_t)gridDim.x * blockDim.x;
    for (; i < n; i += stride) out[i] = (bf16_t)in[i];
}

// ---------------------------------------------------------------------------
// Kernel 3: generic bf16 WMMA GEMM: C[M,N] = A[M,K] @ B[K,N]  (row-major).
// Block tile 128x128, BK=32, 256 threads = 8 waves; each wave a 32x64 subtile
// (2x4 WMMA accumulators). Ping-pong LDS tiles:
//   - A tile staged via GLOBAL_LOAD_ASYNC_TO_LDS_B128 (ASYNCcnt), bypassing
//     VGPRs entirely (register-path fallback if the builtin is absent).
//   - B tile staged transposed so both WMMA operand fragments are contiguous
//     16B ds_load_b128 per lane (CDNA5 16-bit layout: lane-half h -> K in
//     {8h..8h+7, 16+8h..23+8h}).
// One barrier + one s_wait_asynccnt per K-step; next tile always in flight.
// Epilogue (bf16): fp32 acc -> bf16 LDS repack -> global_store_b128.
// All M/N/K are exact tile multiples: no predication, EXEC stays all-ones.
// ---------------------------------------------------------------------------
#define GBM 128
#define GBN 128
#define GBK 32

template <int OUT_F32>
__global__ __launch_bounds__(256)
void pa_gemm_bf16_kernel(const bf16_t* __restrict__ A,
                         const bf16_t* __restrict__ B,
                         bf16_t* __restrict__ Cb,
                         float*  __restrict__ Cf,
                         int M, int N, int K)
{
    struct Tiles {
        bf16_t As[GBM][GBK];   // 8 KB
        bf16_t BsT[GBN][GBK];  // 8 KB
    };
    union SharedU {
        Tiles  tb[2];              // 32 KB ping-pong
        bf16_t stage[8][32][64];   // 32 KB (epilogue repack)
    };
    __shared__ alignas(16) SharedU sh;

    const int t    = threadIdx.x;
    const int lane = t & 31;
    const int wid  = t >> 5;
    const int wm   = wid & 3;      // 0..3 -> 32-row slab
    const int wn   = wid >> 2;     // 0..1 -> 64-col slab
    const int hl   = lane >> 4;    // lane half (K-offset selector)
    const int ln16 = lane & 15;

    const int m0 = blockIdx.y * GBM;
    const int n0 = blockIdx.x * GBN;

    v8f zacc = {};
    v8f acc[2][4];
#pragma unroll
    for (int i = 0; i < 2; ++i)
#pragma unroll
        for (int j = 0; j < 4; ++j) acc[i][j] = zacc;

    // ---- stage tile 0
#if !HAVE_ASYNC_LDS
    v8bf areg[2];
#endif
    v8bf breg[2];
#pragma unroll
    for (int it = 0; it < 2; ++it) {
        const int c = t + it * 256;               // A: 512 x 16B chunks
#if HAVE_ASYNC_LDS
        pa_async_copy_b128(A + (size_t)(m0 + (c >> 2)) * K + (c & 3) * 8,
                           &sh.tb[0].As[c >> 2][(c & 3) * 8]);
#else
        areg[it] = *(const v8bf*)(A + (size_t)(m0 + (c >> 2)) * K + (c & 3) * 8);
#endif
        breg[it] = *(const v8bf*)(B + (size_t)(c >> 4) * N + n0 + (c & 15) * 8);
    }

    const int NT = K / GBK;
    for (int i = 0; i < NT; ++i) {
        const int cur = i & 1;
        // ---- B regs -> LDS (transposed); A regs -> LDS on fallback path
#pragma unroll
        for (int it = 0; it < 2; ++it) {
            const int c  = t + it * 256;
            const int kk = c >> 4, ch = c & 15;
#pragma unroll
            for (int j = 0; j < 8; ++j) sh.tb[cur].BsT[ch * 8 + j][kk] = breg[it][j];
#if !HAVE_ASYNC_LDS
            *(v8bf*)(&sh.tb[cur].As[c >> 2][(c & 3) * 8]) = areg[it];
#endif
        }
        pa_async_wait0();          // A tile `i` landed in LDS (async path)
        __syncthreads();

        // ---- stage tile i+1 (overlaps the WMMAs below)
        if (i + 1 < NT) {
            const int k1 = (i + 1) * GBK;
            const int nxt = cur ^ 1;
#pragma unroll
            for (int it = 0; it < 2; ++it) {
                const int c = t + it * 256;
#if HAVE_ASYNC_LDS
                pa_async_copy_b128(A + (size_t)(m0 + (c >> 2)) * K + k1 + (c & 3) * 8,
                                   &sh.tb[nxt].As[c >> 2][(c & 3) * 8]);
#else
                areg[it] = *(const v8bf*)(A + (size_t)(m0 + (c >> 2)) * K + k1 + (c & 3) * 8);
#endif
                breg[it] = *(const v8bf*)(B + (size_t)(k1 + (c >> 4)) * N + n0 + (c & 15) * 8);
            }
            (void)nxt;
        }

        // ---- WMMA compute
        Frag16 afr[2], bfr[4];
#pragma unroll
        for (int mt = 0; mt < 2; ++mt) {
            const int row = wm * 32 + mt * 16 + ln16;
            afr[mt].h[0] = *(const v8bf*)(&sh.tb[cur].As[row][8 * hl]);
            afr[mt].h[1] = *(const v8bf*)(&sh.tb[cur].As[row][16 + 8 * hl]);
        }
#pragma unroll
        for (int nt = 0; nt < 4; ++nt) {
            const int col = wn * 64 + nt * 16 + ln16;
            bfr[nt].h[0] = *(const v8bf*)(&sh.tb[cur].BsT[col][8 * hl]);
            bfr[nt].h[1] = *(const v8bf*)(&sh.tb[cur].BsT[col][16 + 8 * hl]);
        }
#pragma unroll
        for (int mt = 0; mt < 2; ++mt)
#pragma unroll
            for (int nt = 0; nt < 4; ++nt)
                acc[mt][nt] = wmma_bf16(afr[mt].v, bfr[nt].v, acc[mt][nt]);
    }

    // ---- epilogue. C layout: VGPR r, lane L -> M = r + 8*(L/16), N = L%16
    if (OUT_F32) {
        // small GEMM only (final projection): direct f32 stores, base+offset math
        float* cbase = Cf + (size_t)(m0 + wm * 32 + 8 * hl) * N
                          + (n0 + wn * 64 + ln16);
#pragma unroll
        for (int mt = 0; mt < 2; ++mt)
#pragma unroll
            for (int r = 0; r < 8; ++r) {
                float* crow = cbase + (size_t)(mt * 16 + r) * N;
#pragma unroll
                for (int nt = 0; nt < 4; ++nt)
                    crow[nt * 16] = acc[mt][nt][r];
            }
    } else {
        __syncthreads();   // done with tiles; reuse LDS as repack stage
#pragma unroll
        for (int mt = 0; mt < 2; ++mt)
#pragma unroll
            for (int nt = 0; nt < 4; ++nt)
#pragma unroll
                for (int r = 0; r < 8; ++r)
                    sh.stage[wid][mt * 16 + r + 8 * hl][nt * 16 + ln16] =
                        (bf16_t)acc[mt][nt][r];
        __syncthreads();
        // each wave streams its own 32x64 bf16 subtile: 8 x global_store_b128
        bf16_t* cwave = Cb + (size_t)(m0 + wm * 32) * N + n0 + wn * 64;
#pragma unroll
        for (int i = 0; i < 8; ++i) {
            const int c   = lane + i * 32;   // 256 chunks of 16B
            const int row = c >> 3;
            const int ch  = c & 7;
            *(v8bf*)(cwave + (size_t)row * N + ch * 8) =
                *(const v8bf*)(&sh.stage[wid][row][ch * 8]);
        }
    }
}

// ---------------------------------------------------------------------------
// Kernel 4: flash attention. One block (128 thr = 4 waves) per (batch, head).
// Wave w owns query rows w*16..w*16+15. S=4160 streamed in chunks of 64 with
// register prefetch of the next k/v chunk overlapping the WMMA phase.
// Faithful to reference: scores = q@k^T (no pre-scale), row-max subtracted,
// softmax, THEN /sqrt(d)=8 folded into the final 1/(l*8) normalization.
// ---------------------------------------------------------------------------
__global__ __launch_bounds__(128)
void pa_flash_attn_kernel(const bf16_t* __restrict__ qbuf,   // [B*64, 1024]
                          const bf16_t* __restrict__ kv,     // [B*4160, 2048]
                          bf16_t* __restrict__ attnout)      // [B*64, 1024]
{
    const int bh   = blockIdx.x;
    const int b    = bh >> 4;
    const int head = bh & 15;

    __shared__ alignas(16) bf16_t kt[64][64];        // k chunk, row-major [s][d]
    __shared__ alignas(16) bf16_t vT[64][64];        // v chunk, transposed [d][s]
    __shared__ alignas(16) bf16_t pt[4][16][64];     // per-wave P tile [m][s]

    const int t    = threadIdx.x;
    const int lane = t & 31;
    const int w    = t >> 5;
    const int hl   = lane >> 4;
    const int ln16 = lane & 15;

    // q A-fragments (held in registers for the whole kernel)
    Frag16 aq[2];
    {
        const int lq = w * 16 + ln16;
        const bf16_t* qrow = qbuf + ((size_t)b * N_LAT + lq) * HIDDEN_ + head * DHEAD_;
#pragma unroll
        for (int kk = 0; kk < 2; ++kk) {
            aq[kk].h[0] = *(const v8bf*)(qrow + kk * 32 + 8 * hl);
            aq[kk].h[1] = *(const v8bf*)(qrow + kk * 32 + 16 + 8 * hl);
        }
    }

    v8f zacc = {};
    v8f O[4];
#pragma unroll
    for (int dt = 0; dt < 4; ++dt) O[dt] = zacc;
    float m_run[8], l_run[8];
#pragma unroll
    for (int r = 0; r < 8; ++r) { m_run[r] = -1.0e30f; l_run[r] = 0.f; }

    const bf16_t* kv_bh = kv + (size_t)b * CTX_LEN * (2 * HIDDEN_) + head * DHEAD_;

    // ---- prefetch chunk s0=0 (each thread: 4 k-chunks + 4 v-chunks of 16B)
    v8bf kreg[4], vreg[4];
#pragma unroll
    for (int it = 0; it < 4; ++it) {
        const int c = t + it * 128;             // 512 chunks of 16B
        const bf16_t* base = kv_bh + (size_t)(c >> 3) * (2 * HIDDEN_) + (c & 7) * 8;
        kreg[it] = *(const v8bf*)(base);
        vreg[it] = *(const v8bf*)(base + HIDDEN_);
    }

    for (int s0 = 0; s0 < CTX_LEN; s0 += 64) {
        __syncthreads();
        // ---- registers -> LDS (k row-major, v transposed)
#pragma unroll
        for (int it = 0; it < 4; ++it) {
            const int c   = t + it * 128;
            const int row = c >> 3;
            const int ch  = c & 7;
            *(v8bf*)(&kt[row][ch * 8]) = kreg[it];
#pragma unroll
            for (int j = 0; j < 8; ++j) vT[ch * 8 + j][row] = vreg[it][j];
        }
        __syncthreads();

        // ---- prefetch next chunk (overlaps the WMMAs below)
        const int s1 = s0 + 64;
        if (s1 < CTX_LEN) {
#pragma unroll
            for (int it = 0; it < 4; ++it) {
                const int c = t + it * 128;
                const bf16_t* base =
                    kv_bh + (size_t)(s1 + (c >> 3)) * (2 * HIDDEN_) + (c & 7) * 8;
                kreg[it] = *(const v8bf*)(base);
                vreg[it] = *(const v8bf*)(base + HIDDEN_);
            }
        }

        // ---- scores S[16 q x 64 s] per wave
        v8f sc[4];
#pragma unroll
        for (int nt = 0; nt < 4; ++nt) {
            sc[nt] = zacc;
            const int srow = nt * 16 + ln16;   // lane -> s column
            Frag16 bk;
#pragma unroll
            for (int kk = 0; kk < 2; ++kk) {
                bk.h[0] = *(const v8bf*)(&kt[srow][kk * 32 + 8 * hl]);
                bk.h[1] = *(const v8bf*)(&kt[srow][kk * 32 + 16 + 8 * hl]);
                sc[nt] = wmma_bf16(aq[kk].v, bk.v, sc[nt]);
            }
        }

        // ---- online softmax: 8 rows per lane-half; 16-lane shfl_xor reduce
        float alpha[8];
#pragma unroll
        for (int r = 0; r < 8; ++r) {
            float pm = fmaxf(fmaxf(sc[0][r], sc[1][r]), fmaxf(sc[2][r], sc[3][r]));
#pragma unroll
            for (int msk = 8; msk >= 1; msk >>= 1)
                pm = fmaxf(pm, __shfl_xor(pm, msk, 32));
            const float mn = fmaxf(m_run[r], pm);
            alpha[r] = __expf(m_run[r] - mn);
            float rsum = 0.f;
#pragma unroll
            for (int nt = 0; nt < 4; ++nt) {
                const float e = __expf(sc[nt][r] - mn);
                sc[nt][r] = e;
                rsum += e;
            }
#pragma unroll
            for (int msk = 8; msk >= 1; msk >>= 1)
                rsum += __shfl_xor(rsum, msk, 32);
            l_run[r] = l_run[r] * alpha[r] + rsum;
            m_run[r] = mn;
        }

        // ---- rescale O, spill P (bf16) to per-wave LDS tile
#pragma unroll
        for (int dt = 0; dt < 4; ++dt)
#pragma unroll
            for (int r = 0; r < 8; ++r) O[dt][r] *= alpha[r];
#pragma unroll
        for (int nt = 0; nt < 4; ++nt)
#pragma unroll
            for (int r = 0; r < 8; ++r)
                pt[w][r + 8 * hl][nt * 16 + ln16] = (bf16_t)sc[nt][r];
        __syncthreads();

        // ---- O += P @ V
        Frag16 ap[2];
#pragma unroll
        for (int kk = 0; kk < 2; ++kk) {
            ap[kk].h[0] = *(const v8bf*)(&pt[w][ln16][kk * 32 + 8 * hl]);
            ap[kk].h[1] = *(const v8bf*)(&pt[w][ln16][kk * 32 + 16 + 8 * hl]);
        }
#pragma unroll
        for (int dt = 0; dt < 4; ++dt) {
            const int dcol = dt * 16 + ln16;
            Frag16 bv;
#pragma unroll
            for (int kk = 0; kk < 2; ++kk) {
                bv.h[0] = *(const v8bf*)(&vT[dcol][kk * 32 + 8 * hl]);
                bv.h[1] = *(const v8bf*)(&vT[dcol][kk * 32 + 16 + 8 * hl]);
                O[dt] = wmma_bf16(ap[kk].v, bv.v, O[dt]);
            }
        }
    }

    // ---- normalize by l and post-softmax /sqrt(d)=8, store bf16
    bf16_t* obase = attnout + ((size_t)b * N_LAT + w * 16 + 8 * hl) * HIDDEN_
                            + head * DHEAD_ + ln16;
#pragma unroll
    for (int r = 0; r < 8; ++r) {
        const float inv = 1.0f / (l_run[r] * 8.0f);
        bf16_t* orow = obase + (size_t)r * HIDDEN_;
#pragma unroll
        for (int dt = 0; dt < 4; ++dt)
            orow[dt * 16] = (bf16_t)(O[dt][r] * inv);
    }
}

// ---------------------------------------------------------------------------
// Host launcher
// ---------------------------------------------------------------------------
extern "C" void kernel_launch(void* const* d_in, const int* in_sizes, int n_in,
                              void* d_out, int out_size, void* d_ws, size_t ws_size,
                              hipStream_t stream)
{
    (void)in_sizes; (void)n_in; (void)out_size; (void)ws_size;

    const float* x       = (const float*)d_in[0];
    const float* latents = (const float*)d_in[1];
    const float* g_media = (const float*)d_in[2];
    const float* b_media = (const float*)d_in[3];
    const float* g_lat   = (const float*)d_in[4];
    const float* b_lat   = (const float*)d_in[5];
    const float* Wq      = (const float*)d_in[6];
    const float* Wkv     = (const float*)d_in[7];
    const float* Wout    = (const float*)d_in[8];
    float* out = (float*)d_out;

    char* ws = (char*)d_ws;
    size_t off = 0;
    auto alloc = [&](size_t bytes) -> char* {
        char* p = ws + off;
        off += (bytes + 255) & ~(size_t)255;
        return p;
    };

    const size_t MCTX = (size_t)B_SZ * CTX_LEN;            // 33280
    bf16_t* ctx    = (bf16_t*)alloc(MCTX * DIM_ * 2);          // 68.2 MB
    bf16_t* lnc    = (bf16_t*)alloc((size_t)B_SZ * N_LAT * DIM_ * 2);
    bf16_t* wq_b   = (bf16_t*)alloc((size_t)DIM_ * HIDDEN_ * 2);
    bf16_t* wkv_b  = (bf16_t*)alloc((size_t)DIM_ * 2 * HIDDEN_ * 2);
    bf16_t* wout_b = (bf16_t*)alloc((size_t)HIDDEN_ * DIM_ * 2);
    bf16_t* qb     = (bf16_t*)alloc((size_t)B_SZ * N_LAT * HIDDEN_ * 2);
    bf16_t* kvb    = (bf16_t*)alloc(MCTX * 2 * HIDDEN_ * 2);   // 136.3 MB
    bf16_t* ao     = (bf16_t*)alloc((size_t)B_SZ * N_LAT * HIDDEN_ * 2);

    // 1. LayerNorm(x) + LayerNorm(latents) -> bf16 ctx (concat fused) + lnc
    pa_layernorm_kernel<<<(int)MCTX, 128, 0, stream>>>(
        x, latents, g_media, b_media, g_lat, b_lat, ctx, lnc);

    // 2. weight conversion fp32 -> bf16
    pa_f32_to_bf16_kernel<<<1024, 256, 0, stream>>>(Wq,   wq_b,   (size_t)DIM_ * HIDDEN_);
    pa_f32_to_bf16_kernel<<<2048, 256, 0, stream>>>(Wkv,  wkv_b,  (size_t)DIM_ * 2 * HIDDEN_);
    pa_f32_to_bf16_kernel<<<1024, 256, 0, stream>>>(Wout, wout_b, (size_t)HIDDEN_ * DIM_);

    // 3. q = lnc @ Wq   [512,1024] = [512,1024]x[1024,1024]
    pa_gemm_bf16_kernel<0><<<dim3(HIDDEN_ / GBN, (B_SZ * N_LAT) / GBM), 256, 0, stream>>>(
        lnc, wq_b, qb, nullptr, B_SZ * N_LAT, HIDDEN_, DIM_);

    // 4. kv = ctx @ Wkv  [33280,2048] = [33280,1024]x[1024,2048]  (dominant GEMM)
    pa_gemm_bf16_kernel<0><<<dim3((2 * HIDDEN_) / GBN, (int)(MCTX / GBM)), 256, 0, stream>>>(
        ctx, wkv_b, kvb, nullptr, (int)MCTX, 2 * HIDDEN_, DIM_);

    // 5. fused attention (softmax never hits HBM)
    pa_flash_attn_kernel<<<B_SZ * HEADS_, 128, 0, stream>>>(qb, kvb, ao);

    // 6. out = ao @ Wout  -> fp32 d_out [512,1024]
    pa_gemm_bf16_kernel<1><<<dim3(DIM_ / GBN, (B_SZ * N_LAT) / GBM), 256, 0, stream>>>(
        ao, wout_b, nullptr, out, B_SZ * N_LAT, DIM_, HIDDEN_);
}
